// ASEBlock_53781580481024
// MI455X (gfx1250) — matrix operand order repaired and checked by reading, not compiled, not run on hardware
//
#include <hip/hip_runtime.h>
#include <math.h>

// ---------------------------------------------------------------------------
// Types for CDNA5 WMMA (wave32). bf16 A/B, f32 accumulate.
// ---------------------------------------------------------------------------
typedef __attribute__((ext_vector_type(16))) __bf16 v16bf;
typedef __attribute__((ext_vector_type(8)))  __bf16 bf16x8;
typedef __attribute__((ext_vector_type(8)))  float  v8f;

#define DEVFN __device__ __forceinline__

DEVFN float gelu_f(float v) { return 0.5f * v * (1.0f + erff(v * 0.70710678118f)); }
DEVFN float silu_f(float v) { return v / (1.0f + expf(-v)); }

// Fixed problem dims (from reference): B=8, Cin=64, H=W=64 post-pool, C=128.
#define NB   8
#define HH   64
#define WW   64
#define CC   128
#define CIN  64
#define MM   (NB * HH * WW)      // 32768 pixel rows
#define LL   (HH * WW)           // 4096 sequence length

// ---------------------------------------------------------------------------
// Max-pool 2x2 on (8,64,128,128) NCHW f32 -> (8,64,64,64,Cin) NHWC bf16
// ---------------------------------------------------------------------------
__global__ void pool_kernel(const float* __restrict__ x, __bf16* __restrict__ xp) {
  int t = blockIdx.x * blockDim.x + threadIdx.x;
  if (t >= NB * HH * WW * CIN) return;
  int c = t & 63;
  int r = t >> 6;
  int xc = r & 63; r >>= 6;
  int yc = r & 63;
  int b  = r >> 6;
  const float* p = x + (((size_t)(b * CIN + c) * 128 + yc * 2) * 128 + xc * 2);
  float m0 = fmaxf(p[0], p[1]);
  float m1 = fmaxf(p[128], p[129]);
  xp[t] = (__bf16)fmaxf(m0, m1);
}

// ---------------------------------------------------------------------------
// Pack weights OIHW f32 -> [kk][O][I] bf16 (I contiguous: matches WMMA B frag)
// ---------------------------------------------------------------------------
__global__ void pack_w_kernel(const float* __restrict__ w, __bf16* __restrict__ out,
                              int O, int I, int KHW) {
  int t = blockIdx.x * blockDim.x + threadIdx.x;
  if (t >= O * I * KHW) return;
  int i = t % I; int rest = t / I;
  int o = rest % O; int kk = rest / O;
  out[((size_t)kk * O + o) * I + i] = (__bf16)w[((size_t)o * I + i) * KHW + kk];
}

// ---------------------------------------------------------------------------
// Fold BN (+bias) into per-channel scale/shift:
//   y = conv*scale + shift, scale=g/sqrt(v+eps), shift=(bias-mean)*scale+beta
// ---------------------------------------------------------------------------
__global__ void bnfold_kernel(const float* __restrict__ g, const float* __restrict__ b,
                              const float* __restrict__ m, const float* __restrict__ v,
                              const float* __restrict__ bias,
                              float* __restrict__ scale, float* __restrict__ shift, int Cn) {
  int c = blockIdx.x * blockDim.x + threadIdx.x;
  if (c >= Cn) return;
  float s = g[c] * rsqrtf(v[c] + 1e-5f);
  scale[c] = s;
  shift[c] = (bias[c] - m[c]) * s + b[c];
}

// ---------------------------------------------------------------------------
// Implicit-GEMM convolution via V_WMMA_F32_16X16X32_BF16, fully templated.
//   A: NHWC bf16 activations (NB,H,W,TCIN); Wp: packed bf16 [KH*KW][128][TCIN]
// One wave computes a 16-pixel x 64-outchannel tile: one shared A fragment,
// 4 B fragments, 4 f32 accumulators (Cout fixed at 128 -> 2 channel groups).
// Fragment maps (ISA 7.12.2, wave32):
//   A (16-bit 16x32): lane<16 -> K0..15 ; lane>=16 -> K16..31 (this code's order, not the ISA table)
//   B (16-bit 32x16): lane n -> 16 contiguous K of column n (+16 for lane>=16), this code's order, same as A
//   D (f32 16x16):    vgpr r: lanes0-15 M=r, lanes16-31 M=8+r
// ---------------------------------------------------------------------------
template <int TCIN, int TKH, int TKW, int TPAD, int TACT>
__global__ __launch_bounds__(256)
void conv_wmma_kernel(const __bf16* __restrict__ A, const __bf16* __restrict__ Wp,
                      const float* __restrict__ scale, const float* __restrict__ shift,
                      float* __restrict__ outF, __bf16* __restrict__ outB) {
  int wid  = (int)((blockIdx.x * (size_t)blockDim.x + threadIdx.x) >> 5);
  int lane = threadIdx.x & 31;
  int mt = wid >> 1;                 // 0..2047 pixel tiles (M=32768/16)
  int ng = wid & 1;                  // 64-wide channel group
  int rowIdx = mt >> 2;              // b*H + y  (W/16 = 4 tiles per row)
  int x0 = (mt & 3) << 4;
  int y = rowIdx & (HH - 1);
  int imgBase = rowIdx - y;          // b*H
  int n0 = ng << 6;
  int ml  = lane & 15;
  int kgA = (lane >> 4) << 4;        // 0 or 16
  int kgB = (lane >> 4) << 4;        // 0 or 16

  v8f acc[4];
#pragma unroll
  for (int j = 0; j < 4; ++j)
#pragma unroll
    for (int r = 0; r < 8; ++r) acc[j][r] = 0.0f;

  const __bf16* wb = Wp + (size_t)(n0 + ml) * TCIN + kgB;
  constexpr int KHW = TKH * TKW;
  for (int kk = 0; kk < KHW; ++kk) {
    int ky = kk / TKW, kx = kk - ky * TKW;
    int yy = y + ky - TPAD;
    if constexpr (TPAD > 0) {
      if (yy < 0 || yy >= HH) continue;
    }
    int xx = x0 + ml + kx - TPAD;
    bool valid;
    if constexpr (TPAD > 0) valid = (xx >= 0) && (xx < WW);
    else                    valid = true;
    const __bf16* arow = A + ((size_t)(imgBase + yy) * WW + (valid ? xx : 0)) * TCIN;
    const __bf16* wrow = wb + (size_t)kk * (CC * TCIN);
    if constexpr (KHW > 1) {
      if (kk + 1 < KHW)  // stream next tap's weight block into cache
        __builtin_prefetch((const void*)(wrow + (size_t)(CC * TCIN)), 0, 1);
    }
#pragma unroll
    for (int kb = 0; kb < TCIN; kb += 32) {
      v16bf av;
      if (valid) {
        bf16x8 lo = *(const bf16x8*)(arow + kb + kgA);
        bf16x8 hi = *(const bf16x8*)(arow + kb + kgA + 8);
#pragma unroll
        for (int i = 0; i < 8; ++i) { av[i] = lo[i]; av[8 + i] = hi[i]; }
      } else {
#pragma unroll
        for (int i = 0; i < 16; ++i) av[i] = (__bf16)0.0f;
      }
#pragma unroll
      for (int j = 0; j < 4; ++j) {
        v16bf bv = *(const v16bf*)(wrow + (size_t)j * 16 * TCIN + kb);
        acc[j] = __builtin_amdgcn_wmma_f32_16x16x32_bf16(false, av, false, bv,
                                                         (short)0, acc[j], false, false);
      }
    }
  }

  int mo = (lane >> 4) << 3;
  size_t pixBase = (size_t)rowIdx * WW + x0;
#pragma unroll
  for (int j = 0; j < 4; ++j) {
    int ch = n0 + j * 16 + ml;
    float sc = scale ? scale[ch] : 1.0f;
    float sh = shift ? shift[ch] : 0.0f;
#pragma unroll
    for (int r = 0; r < 8; ++r) {
      float v = acc[j][r] * sc + sh;
      if constexpr (TACT == 1) v = fmaxf(v, 0.0f);
      if constexpr (TACT == 2) v = gelu_f(v);
      size_t idx = (pixBase + (size_t)(r + mo)) * CC + ch;
      if (outF) outF[idx] = v;
      if (outB) outB[idx] = (__bf16)v;
    }
  }
}

// ---------------------------------------------------------------------------
// LayerNorm over C=128 (last dim). One wave32 per row; optional residual add
// BEFORE the norm; writes f32 and/or bf16.
// ---------------------------------------------------------------------------
__global__ void ln_kernel(const float* __restrict__ in, const float* __restrict__ res,
                          const float* __restrict__ gamma, const float* __restrict__ beta,
                          float* __restrict__ outF, __bf16* __restrict__ outB, int M) {
  int wid  = (int)((blockIdx.x * (size_t)blockDim.x + threadIdx.x) >> 5);
  int lane = threadIdx.x & 31;
  if (wid >= M) return;
  const float* row = in + (size_t)wid * CC;
  float v[4]; float s = 0.f, s2 = 0.f;
#pragma unroll
  for (int i = 0; i < 4; ++i) {
    float t = row[lane + 32 * i];
    if (res) t += res[(size_t)wid * CC + lane + 32 * i];
    v[i] = t; s += t; s2 += t * t;
  }
#pragma unroll
  for (int o = 16; o > 0; o >>= 1) { s += __shfl_xor(s, o, 32); s2 += __shfl_xor(s2, o, 32); }
  float mu  = s * (1.0f / CC);
  float var = s2 * (1.0f / CC) - mu * mu;
  float inv = rsqrtf(var + 1e-5f);
#pragma unroll
  for (int i = 0; i < 4; ++i) {
    int ch = lane + 32 * i;
    float o = (v[i] - mu) * inv * gamma[ch] + beta[ch];
    if (outF) outF[(size_t)wid * CC + ch] = o;
    if (outB) outB[(size_t)wid * CC + ch] = (__bf16)o;
  }
}

// ---------------------------------------------------------------------------
// Mamba (dm=32, di=64, ds=16, dtr=2, dconv=4), chunk-sequential (co=32*chunk).
// ---------------------------------------------------------------------------
__global__ void m_inproj(const float* __restrict__ tn, const float* __restrict__ wip,
                         float* __restrict__ xcpre, float* __restrict__ z, int co) {
  int t = blockIdx.x * blockDim.x + threadIdx.x;
  if (t >= MM * 128) return;
  int n = t & 127; size_t row = (size_t)(t >> 7);
  const float* a = tn + row * CC + co;
  const float* w = wip + n * 32;
  float acc = 0.f;
#pragma unroll
  for (int k = 0; k < 32; ++k) acc += a[k] * w[k];
  if (n < 64) xcpre[row * 64 + n] = acc;
  else        z[row * 64 + (n - 64)] = acc;
}

__global__ void m_conv(const float* __restrict__ xcpre, const float* __restrict__ cw,
                       const float* __restrict__ cb, float* __restrict__ xc) {
  int t = blockIdx.x * blockDim.x + threadIdx.x;
  if (t >= NB * LL * 64) return;
  int d = t & 63; int l = (t >> 6) & (LL - 1); int b = t >> 18;
  float acc = cb[d];
#pragma unroll
  for (int j = 0; j < 4; ++j) {
    int ll = l - 3 + j;
    if (ll >= 0) acc += xcpre[(((size_t)b << 12) + ll) * 64 + d] * cw[d * 4 + j];
  }
  xc[t] = silu_f(acc);
}

__global__ void m_xdbl(const float* __restrict__ xc, const float* __restrict__ xp,
                       float* __restrict__ xdbl) {
  int t = blockIdx.x * blockDim.x + threadIdx.x;
  if (t >= MM * 34) return;
  int j = t % 34; size_t row = (size_t)(t / 34);
  const float* a = xc + row * 64;
  const float* w = xp + j * 64;
  float acc = 0.f;
#pragma unroll
  for (int k = 0; k < 64; ++k) acc += a[k] * w[k];
  xdbl[row * 34 + j] = acc;
}

__global__ void m_dt(const float* __restrict__ xdbl, const float* __restrict__ dtp,
                     const float* __restrict__ dtb, float* __restrict__ dt) {
  int t = blockIdx.x * blockDim.x + threadIdx.x;
  if (t >= MM * 64) return;
  int d = t & 63; size_t row = (size_t)(t >> 6);
  float v = xdbl[row * 34] * dtp[d * 2] + xdbl[row * 34 + 1] * dtp[d * 2 + 1] + dtb[d];
  dt[t] = (v > 20.f) ? v : log1pf(expf(v));
}

// Sequential selective scan: grid = 8 (batch), block = 64 (d). Gated output is
// written in-place into the z buffer (each element owned by exactly one thread).
__global__ void m_scan(const float* __restrict__ dt, const float* __restrict__ xdbl,
                       const float* __restrict__ xc, const float* __restrict__ Alog,
                       const float* __restrict__ Dp, float* __restrict__ zy) {
  int b = blockIdx.x;
  int d = threadIdx.x;
  __shared__ float sBC[32];
  float Ad[16];
#pragma unroll
  for (int s = 0; s < 16; ++s) Ad[s] = -expf(Alog[d * 16 + s]);
  float Dv = Dp[d];
  float h[16];
#pragma unroll
  for (int s = 0; s < 16; ++s) h[s] = 0.f;
  for (int l = 0; l < LL; ++l) {
    size_t row = ((size_t)b << 12) + l;
    if (d < 32) sBC[d] = xdbl[row * 34 + 2 + d];   // B rows then C rows
    __syncthreads();
    float dtv = dt[row * 64 + d];
    float xv  = xc[row * 64 + d];
    float acc = 0.f;
#pragma unroll
    for (int s = 0; s < 16; ++s) {
      h[s] = expf(dtv * Ad[s]) * h[s] + dtv * sBC[s] * xv;
      acc += h[s] * sBC[16 + s];
    }
    float y  = acc + xv * Dv;
    float zv = zy[row * 64 + d];
    zy[row * 64 + d] = y * silu_f(zv);
    __syncthreads();
  }
}

__global__ void m_outproj(const float* __restrict__ zy, const float* __restrict__ wop,
                          const float* __restrict__ tn, float* __restrict__ xm, int co) {
  int t = blockIdx.x * blockDim.x + threadIdx.x;
  if (t >= MM * 32) return;
  int n = t & 31; size_t row = (size_t)(t >> 5);
  const float* a = zy + row * 64;
  const float* w = wop + n * 64;
  float acc = 0.f;
#pragma unroll
  for (int k = 0; k < 64; ++k) acc += a[k] * w[k];
  xm[row * CC + co + n] = acc + tn[row * CC + co + n];   // + residual chunk
}

// ---------------------------------------------------------------------------
// Depthwise 7x7 (NHWC f32 in) + bias + exact GELU -> bf16
// ---------------------------------------------------------------------------
__global__ void dw7_kernel(const float* __restrict__ f, const float* __restrict__ w,
                           const float* __restrict__ bias, __bf16* __restrict__ outB) {
  int t = blockIdx.x * blockDim.x + threadIdx.x;
  if (t >= MM * CC) return;
  int ch = t & 127; int pix = t >> 7;
  int xc = pix & 63, yc = (pix >> 6) & 63, b = pix >> 12;
  float acc = bias[ch];
  for (int ky = 0; ky < 7; ++ky) {
    int yy = yc + ky - 3; if (yy < 0 || yy >= HH) continue;
    for (int kx = 0; kx < 7; ++kx) {
      int xx = xc + kx - 3; if (xx < 0 || xx >= WW) continue;
      acc += f[((size_t)((b * HH + yy) * WW + xx)) * CC + ch] * w[ch * 49 + ky * 7 + kx];
    }
  }
  outB[t] = (__bf16)gelu_f(acc);
}

// ---------------------------------------------------------------------------
// Offset conv: 3x3, Cout=18 (not WMMA-shaped), NHWC in, fused BN.
// ---------------------------------------------------------------------------
__global__ void offconv_kernel(const float* __restrict__ g, const float* __restrict__ w,
                               const float* __restrict__ scale, const float* __restrict__ shift,
                               float* __restrict__ off) {
  int t = blockIdx.x * blockDim.x + threadIdx.x;
  if (t >= MM * 18) return;
  int oc = t % 18; int pix = t / 18;
  int xc = pix & 63, yc = (pix >> 6) & 63, b = pix >> 12;
  float acc = 0.f;
  for (int ky = 0; ky < 3; ++ky) {
    int yy = yc + ky - 1; if (yy < 0 || yy >= HH) continue;
    for (int kx = 0; kx < 3; ++kx) {
      int xx = xc + kx - 1; if (xx < 0 || xx >= WW) continue;
      const float* grow = g + ((size_t)((b * HH + yy) * WW + xx)) * CC;
      const float* wr = w + ((size_t)oc * CC) * 9 + ky * 3 + kx;
      float a = 0.f;
      for (int ci = 0; ci < CC; ++ci) a += grow[ci] * wr[ci * 9];
      acc += a;
    }
  }
  off[(size_t)pix * 18 + oc] = acc * scale[oc] + shift[oc];
}

// ---------------------------------------------------------------------------
// Deformable depthwise 3x3 with bilinear sampling -> bf16 (input to bal GEMM)
// ---------------------------------------------------------------------------
__global__ void deform_kernel(const float* __restrict__ g, const float* __restrict__ off,
                              const float* __restrict__ dw, __bf16* __restrict__ dB) {
  int t = blockIdx.x * blockDim.x + threadIdx.x;
  if (t >= MM * CC) return;
  int ch = t & 127; int pix = t >> 7;
  int xc = pix & 63, yc = (pix >> 6) & 63, b = pix >> 12;
  const float* gb = g + ((size_t)b * LL) * CC + ch;
  float acc = 0.f;
  for (int kk = 0; kk < 9; ++kk) {
    float py = (float)(yc + kk / 3 - 1) + off[(size_t)pix * 18 + kk * 2];
    float px = (float)(xc + kk % 3 - 1) + off[(size_t)pix * 18 + kk * 2 + 1];
    float y0f = floorf(py), x0f = floorf(px);
    int y0 = (int)y0f, x0 = (int)x0f;
    float wy = py - y0f, wx = px - x0f;
    float v = 0.f;
#pragma unroll
    for (int dy = 0; dy < 2; ++dy)
#pragma unroll
      for (int dx = 0; dx < 2; ++dx) {
        int yi = y0 + dy, xi = x0 + dx;
        float wgt = (dy ? wy : 1.f - wy) * (dx ? wx : 1.f - wx);
        if (yi >= 0 && yi < HH && xi >= 0 && xi < WW)
          v += gb[((size_t)(yi * WW + xi)) * CC] * wgt;
      }
    acc += v * dw[ch * 9 + kk];
  }
  dB[t] = (__bf16)acc;
}

// ---------------------------------------------------------------------------
// Final fuse: out(NCHW) = x_ma + gelu(dbal)*h + h
// ---------------------------------------------------------------------------
__global__ void final_kernel(const float* __restrict__ xma, const float* __restrict__ dbal,
                             const float* __restrict__ hbuf, float* __restrict__ out) {
  int t = blockIdx.x * blockDim.x + threadIdx.x;
  if (t >= NB * CC * HH * WW) return;
  int xx = t & 63, yy = (t >> 6) & 63, ch = (t >> 12) & 127, b = t >> 19;
  size_t pix = ((size_t)b * HH + yy) * WW + xx;
  float hv = hbuf[pix * CC + ch];
  float dv = dbal[pix * CC + ch];
  out[t] = xma[pix * CC + ch] + gelu_f(dv) * hv + hv;
}

// ---------------------------------------------------------------------------
// Host orchestration
// ---------------------------------------------------------------------------
static inline int cdiv(long a, long b) { return (int)((a + b - 1) / b); }

template <int TCIN, int TKH, int TKW, int TPAD, int TACT>
static void launch_conv(const __bf16* A, const __bf16* Wp, const float* scale,
                        const float* shift, float* outF, __bf16* outB,
                        hipStream_t stream) {
  // waves = Mtiles(2048) * channel-groups(2); 8 waves per 256-thread block
  conv_wmma_kernel<TCIN, TKH, TKW, TPAD, TACT>
      <<<512, 256, 0, stream>>>(A, Wp, scale, shift, outF, outB);
}

extern "C" void kernel_launch(void* const* d_in, const int* in_sizes, int n_in,
                              void* d_out, int out_size, void* d_ws, size_t ws_size,
                              hipStream_t stream) {
  (void)in_sizes; (void)n_in; (void)out_size; (void)ws_size;
  const float* X        = (const float*)d_in[0];
  const float* ipc_w1   = (const float*)d_in[1];
  const float* ipc_b1   = (const float*)d_in[2];
  const float* bn1_g    = (const float*)d_in[3];
  const float* bn1_b    = (const float*)d_in[4];
  const float* bn1_m    = (const float*)d_in[5];
  const float* bn1_v    = (const float*)d_in[6];
  const float* ipc_w2   = (const float*)d_in[7];
  const float* ipc_b2   = (const float*)d_in[8];
  const float* bn2_g    = (const float*)d_in[9];
  const float* bn2_b    = (const float*)d_in[10];
  const float* bn2_m    = (const float*)d_in[11];
  const float* bn2_v    = (const float*)d_in[12];
  const float* ipc_w3   = (const float*)d_in[13];
  const float* ipc_b3   = (const float*)d_in[14];
  const float* bn3_g    = (const float*)d_in[15];
  const float* bn3_b    = (const float*)d_in[16];
  const float* bn3_m    = (const float*)d_in[17];
  const float* bn3_v    = (const float*)d_in[18];
  const float* ln_g     = (const float*)d_in[19];
  const float* ln_b     = (const float*)d_in[20];
  const float* m_inp    = (const float*)d_in[21];
  const float* m_cw     = (const float*)d_in[22];
  const float* m_cb     = (const float*)d_in[23];
  const float* m_xp     = (const float*)d_in[24];
  const float* m_dtp    = (const float*)d_in[25];
  const float* m_dtb    = (const float*)d_in[26];
  const float* m_alog   = (const float*)d_in[27];
  const float* m_D      = (const float*)d_in[28];
  const float* m_outp   = (const float*)d_in[29];
  const float* proj_w   = (const float*)d_in[30];
  const float* proj_b   = (const float*)d_in[31];
  const float* pro1_w   = (const float*)d_in[32];
  const float* pro1_b   = (const float*)d_in[33];
  const float* dw_w     = (const float*)d_in[34];
  const float* dw_b     = (const float*)d_in[35];
  const float* pro2_w   = (const float*)d_in[36];
  const float* pro2_b   = (const float*)d_in[37];
  const float* dcbp_w   = (const float*)d_in[38];
  const float* dcbp_b   = (const float*)d_in[39];
  const float* dcbp_g   = (const float*)d_in[40];
  const float* dcbp_bb  = (const float*)d_in[41];
  const float* dcbp_m   = (const float*)d_in[42];
  const float* dcbp_v   = (const float*)d_in[43];
  const float* off_w    = (const float*)d_in[44];
  const float* off_b    = (const float*)d_in[45];
  const float* offbn_g  = (const float*)d_in[46];
  const float* offbn_b  = (const float*)d_in[47];
  const float* offbn_m  = (const float*)d_in[48];
  const float* offbn_v  = (const float*)d_in[49];
  const float* def_w    = (const float*)d_in[50];
  const float* bal_w    = (const float*)d_in[51];
  const float* bal_b    = (const float*)d_in[52];
  const float* balbn_g  = (const float*)d_in[53];
  const float* balbn_b  = (const float*)d_in[54];
  const float* balbn_m  = (const float*)d_in[55];
  const float* balbn_v  = (const float*)d_in[56];
  float* OUT = (float*)d_out;

  // ---- workspace carve-up (bump allocator, 256B aligned) ----
  char* base = (char*)d_ws; size_t off = 0;
  auto alloc = [&](size_t bytes) -> char* {
    char* p = base + off;
    off += (bytes + 255) & ~(size_t)255;
    return p;
  };
  float*  BNBUF  = (float*)alloc(12 * 128 * sizeof(float)); // 6 x (scale,shift)
  __bf16* XP     = (__bf16*)alloc((size_t)MM * CIN * 2);
  __bf16* W1P    = (__bf16*)alloc((size_t)49 * CC * CIN * 2);
  __bf16* W2P    = (__bf16*)alloc((size_t)9 * CC * CC * 2);
  __bf16* W3P    = (__bf16*)alloc((size_t)CC * CC * 2);
  __bf16* WPROJ  = (__bf16*)alloc((size_t)CC * CC * 2);
  __bf16* WPRO1  = (__bf16*)alloc((size_t)CC * CC * 2);
  __bf16* WPRO2  = (__bf16*)alloc((size_t)CC * CC * 2);
  __bf16* WDCBP  = (__bf16*)alloc((size_t)CC * CC * 2);
  __bf16* WBAL   = (__bf16*)alloc((size_t)CC * CC * 2);
  __bf16* BFA    = (__bf16*)alloc((size_t)MM * CC * 2);   // ping
  __bf16* BFB    = (__bf16*)alloc((size_t)MM * CC * 2);   // pong
  float*  HF     = (float*)alloc((size_t)MM * CC * 4);    // h (lives to end)
  float*  TN     = (float*)alloc((size_t)MM * CC * 4);    // ln(t)
  float*  GF     = (float*)alloc((size_t)MM * CC * 4);    // g, later xm-proj
  float*  FF     = (float*)alloc((size_t)MM * CC * 4);    // f (pro1 out)
  float*  DBAL   = (float*)alloc((size_t)MM * CC * 4);    // bal conv out
  float*  XM     = (float*)alloc((size_t)MM * CC * 4);    // xm raw, later x_ma
  float*  OFF    = (float*)alloc((size_t)MM * 18 * 4);
  float*  XDBL   = (float*)alloc((size_t)MM * 34 * 4);
  float*  XCPRE  = (float*)alloc((size_t)MM * 64 * 4);    // reused as dt
  float*  XC     = (float*)alloc((size_t)MM * 64 * 4);
  float*  ZB     = (float*)alloc((size_t)MM * 64 * 4);    // z, then gated y

  float* sc_bn1  = BNBUF + 0 * 128; float* sh_bn1  = BNBUF + 1 * 128;
  float* sc_bn2  = BNBUF + 2 * 128; float* sh_bn2  = BNBUF + 3 * 128;
  float* sc_bn3  = BNBUF + 4 * 128; float* sh_bn3  = BNBUF + 5 * 128;
  float* sc_dcbp = BNBUF + 6 * 128; float* sh_dcbp = BNBUF + 7 * 128;
  float* sc_off  = BNBUF + 8 * 128; float* sh_off  = BNBUF + 9 * 128;
  float* sc_bal  = BNBUF + 10 * 128; float* sh_bal = BNBUF + 11 * 128;

  // ---- BN folds ----
  bnfold_kernel<<<1, 128, 0, stream>>>(bn1_g, bn1_b, bn1_m, bn1_v, ipc_b1, sc_bn1, sh_bn1, CC);
  bnfold_kernel<<<1, 128, 0, stream>>>(bn2_g, bn2_b, bn2_m, bn2_v, ipc_b2, sc_bn2, sh_bn2, CC);
  bnfold_kernel<<<1, 128, 0, stream>>>(bn3_g, bn3_b, bn3_m, bn3_v, ipc_b3, sc_bn3, sh_bn3, CC);
  bnfold_kernel<<<1, 128, 0, stream>>>(dcbp_g, dcbp_bb, dcbp_m, dcbp_v, dcbp_b, sc_dcbp, sh_dcbp, CC);
  bnfold_kernel<<<1, 128, 0, stream>>>(offbn_g, offbn_b, offbn_m, offbn_v, off_b, sc_off, sh_off, 18);
  bnfold_kernel<<<1, 128, 0, stream>>>(balbn_g, balbn_b, balbn_m, balbn_v, bal_b, sc_bal, sh_bal, CC);

  // ---- pool + weight packs ----
  pool_kernel<<<cdiv((long)MM * CIN, 256), 256, 0, stream>>>(X, XP);
  pack_w_kernel<<<cdiv((long)CC * CIN * 49, 256), 256, 0, stream>>>(ipc_w1, W1P, CC, CIN, 49);
  pack_w_kernel<<<cdiv((long)CC * CC * 9, 256), 256, 0, stream>>>(ipc_w2, W2P, CC, CC, 9);
  pack_w_kernel<<<cdiv((long)CC * CC, 256), 256, 0, stream>>>(ipc_w3, W3P, CC, CC, 1);
  pack_w_kernel<<<cdiv((long)CC * CC, 256), 256, 0, stream>>>(proj_w, WPROJ, CC, CC, 1);
  pack_w_kernel<<<cdiv((long)CC * CC, 256), 256, 0, stream>>>(pro1_w, WPRO1, CC, CC, 1);
  pack_w_kernel<<<cdiv((long)CC * CC, 256), 256, 0, stream>>>(pro2_w, WPRO2, CC, CC, 1);
  pack_w_kernel<<<cdiv((long)CC * CC, 256), 256, 0, stream>>>(dcbp_w, WDCBP, CC, CC, 1);
  pack_w_kernel<<<cdiv((long)CC * CC, 256), 256, 0, stream>>>(bal_w, WBAL, CC, CC, 1);

  // ---- conv stem (WMMA) ----
  launch_conv<CIN, 7, 7, 3, 1>(XP,  W1P, sc_bn1, sh_bn1, nullptr, BFA, stream); // h1
  launch_conv<CC,  3, 3, 1, 1>(BFA, W2P, sc_bn2, sh_bn2, nullptr, BFB, stream); // h2
  launch_conv<CC,  1, 1, 0, 1>(BFB, W3P, sc_bn3, sh_bn3, HF,      BFA, stream); // h

  // ---- deformable branch (runs early to free h-bf16) ----
  launch_conv<CC, 1, 1, 0, 2>(BFA, WDCBP, sc_dcbp, sh_dcbp, GF, nullptr, stream); // g
  offconv_kernel<<<cdiv((long)MM * 18, 256), 256, 0, stream>>>(GF, off_w, sc_off, sh_off, OFF);
  deform_kernel<<<cdiv((long)MM * CC, 256), 256, 0, stream>>>(GF, OFF, def_w, BFB);
  launch_conv<CC, 1, 1, 0, 0>(BFB, WBAL, sc_bal, sh_bal, DBAL, nullptr, stream);  // bn(bal(d))

  // ---- transformer-ish branch ----
  ln_kernel<<<cdiv((long)MM * 32, 256), 256, 0, stream>>>(HF, nullptr, ln_g, ln_b, TN, nullptr, MM); // tn

  for (int c = 0; c < 4; ++c) {
    int co = 32 * c;
    m_inproj<<<cdiv((long)MM * 128, 256), 256, 0, stream>>>(TN, m_inp, XCPRE, ZB, co);
    m_conv<<<cdiv((long)MM * 64, 256), 256, 0, stream>>>(XCPRE, m_cw, m_cb, XC);
    m_xdbl<<<cdiv((long)MM * 34, 256), 256, 0, stream>>>(XC, m_xp, XDBL);
    m_dt<<<cdiv((long)MM * 64, 256), 256, 0, stream>>>(XDBL, m_dtp, m_dtb, XCPRE); // dt aliases xcpre
    m_scan<<<NB, 64, 0, stream>>>(XCPRE, XDBL, XC, m_alog, m_D, ZB);               // gated y in ZB
    m_outproj<<<cdiv((long)MM * 32, 256), 256, 0, stream>>>(ZB, m_outp, TN, XM, co);
  }

  ln_kernel<<<cdiv((long)MM * 32, 256), 256, 0, stream>>>(XM, nullptr, ln_g, ln_b, nullptr, BFA, MM); // ln(xm)
  launch_conv<CC, 1, 1, 0, 0>(BFA, WPROJ, nullptr, proj_b, GF, nullptr, stream);                      // proj
  ln_kernel<<<cdiv((long)MM * 32, 256), 256, 0, stream>>>(GF, HF, ln_g, ln_b, nullptr, BFB, MM);      // xs=ln(.+t)
  launch_conv<CC, 1, 1, 0, 0>(BFB, WPRO1, nullptr, pro1_b, FF, nullptr, stream);                      // f
  dw7_kernel<<<cdiv((long)MM * CC, 256), 256, 0, stream>>>(FF, dw_w, dw_b, BFA);                      // gelu(dw(f))
  launch_conv<CC, 1, 1, 0, 0>(BFA, WPRO2, nullptr, pro2_b, XM, nullptr, stream);                      // x_ma

  // ---- fuse + NHWC->NCHW ----
  final_kernel<<<cdiv((long)NB * CC * HH * WW, 256), 256, 0, stream>>>(XM, DBAL, HF, OUT);
}
